// Mask2FormerDecoderLayer_21191368639162
// MI455X (gfx1250) — matrix operand order repaired and checked by reading, not compiled, not run on hardware
//
#include <hip/hip_runtime.h>
#include <math.h>

// ---------------------------------------------------------------------------
// Mask2Former decoder layer for gfx1250 (MI455X).
// All GEMMs + attention matmuls use v_wmma_f32_16x16x32_bf16 (f32 accum).
// Weights pre-converted to bf16; all WMMA fragments loaded as b128.
// Mask pre-packed to 1 bit/element; wave-private LDS staging (no s_barrier
// in the flash-attention inner loop).
// ---------------------------------------------------------------------------

typedef __attribute__((ext_vector_type(16))) __bf16 v16bf;
typedef __attribute__((ext_vector_type(8)))  float  v8f;

static constexpr int kB      = 8;
static constexpr int kN      = 100;
static constexpr int kDIM    = 256;
static constexpr int kHEADS  = 8;
static constexpr int kHD     = 32;
static constexpr int kFFN    = 2048;
static constexpr int kM      = 16384;     // H*W
static constexpr int kMW     = kM / 32;   // 512 mask words per row
static constexpr int kBN     = kB * kN;   // 800
static constexpr int kBM     = kB * kM;   // 131072
static constexpr float kSCALE = 0.17677669529663687f; // 32^-0.5

__device__ __forceinline__ unsigned short f2bf(float f) {
  unsigned u = __builtin_bit_cast(unsigned, f);
  unsigned r = u + 0x7FFFu + ((u >> 16) & 1u);
  return (unsigned short)(r >> 16);
}
struct bf16x16_raw { uint4 lo, hi; };
__device__ __forceinline__ v16bf make_v16bf(uint4 lo, uint4 hi) {
  bf16x16_raw r{lo, hi};
  return __builtin_bit_cast(v16bf, r);
}
// A fragment (16-bit 16x32, ISA 7.12.2): lane reads ushorts
// [half*8, half*8+8) and [16+half*8, 16+half*8+8)  -> two b128 loads.
__device__ __forceinline__ v16bf load_afrag(const unsigned short* p, int half) {
  uint4 lo = *reinterpret_cast<const uint4*>(p + half * 8);
  uint4 hi = *reinterpret_cast<const uint4*>(p + 16 + half * 8);
  return make_v16bf(lo, hi);
}
// B fragment (16-bit 32x16): lane reads ushorts [half*8, half*8+8) and
// [16+half*8, 16+half*8+8) (same k order as A).
__device__ __forceinline__ v16bf load_bfrag(const unsigned short* p, int half) {
  uint4 lo = *reinterpret_cast<const uint4*>(p + half * 8);
  uint4 hi = *reinterpret_cast<const uint4*>(p + 16 + half * 8);
  return make_v16bf(lo, hi);
}

// ---------------------------------------------------------------------------
// f32 -> bf16 conversion
// ---------------------------------------------------------------------------
__global__ __launch_bounds__(256) void cvt_bf16_kernel(
    const float* __restrict__ in, unsigned short* __restrict__ out, size_t n) {
  size_t i = (size_t)blockIdx.x * blockDim.x + threadIdx.x;
  size_t stride = (size_t)gridDim.x * blockDim.x;
  for (; i < n; i += stride) out[i] = f2bf(in[i]);
}

// ---------------------------------------------------------------------------
// Pack int32 mask -> 1 bit per element. One thread per 32-bit output word
// (reads 32 consecutive ints -> 8 b128 loads).
// ---------------------------------------------------------------------------
__global__ __launch_bounds__(256) void pack_mask_kernel(
    const int* __restrict__ mask, unsigned* __restrict__ mwords, int nwords) {
  int i = blockIdx.x * blockDim.x + threadIdx.x;
  int stride = gridDim.x * blockDim.x;
  for (; i < nwords; i += stride) {
    const int* p = mask + (size_t)i * 32;
    unsigned w = 0;
#pragma unroll
    for (int j = 0; j < 32; ++j) w |= (p[j] != 0 ? 1u : 0u) << j;
    mwords[i] = w;
  }
}

// ---------------------------------------------------------------------------
// LayerNorm over DIM=256: one wave per row.
// ---------------------------------------------------------------------------
__global__ __launch_bounds__(128) void ln_kernel(
    const float* __restrict__ x, const float* __restrict__ g,
    const float* __restrict__ beta, float* __restrict__ outF,
    unsigned short* __restrict__ outB, int rows) {
  int lane = threadIdx.x & 31;
  int wave = threadIdx.x >> 5;
  int row = blockIdx.x * (blockDim.x >> 5) + wave;
  if (row >= rows) return;
  const float* xp = x + (size_t)row * kDIM;
  float v[8];
  float s = 0.f;
#pragma unroll
  for (int i = 0; i < 8; ++i) { v[i] = xp[lane + 32 * i]; s += v[i]; }
#pragma unroll
  for (int m = 16; m; m >>= 1) s += __shfl_xor(s, m, 32);
  float mu = s * (1.0f / 256.0f);
  float vs = 0.f;
#pragma unroll
  for (int i = 0; i < 8; ++i) { float d = v[i] - mu; vs += d * d; }
#pragma unroll
  for (int m = 16; m; m >>= 1) vs += __shfl_xor(vs, m, 32);
  float rstd = rsqrtf(vs * (1.0f / 256.0f) + 1e-5f);
#pragma unroll
  for (int i = 0; i < 8; ++i) {
    int col = lane + 32 * i;
    float y = (v[i] - mu) * rstd * g[col] + beta[col];
    if (outF) outF[(size_t)row * kDIM + col] = y;
    if (outB) outB[(size_t)row * kDIM + col] = f2bf(y);
  }
}

// ---------------------------------------------------------------------------
// Generic GEMM: C[Mr,Nc] = epi((A_bf16[Mr,K] @ Wb_bf16[Nc,K]^T + bias)*scale)
// One wave computes a 16x64 stripe (4 N-tiles share one A fragment).
// Mr, Nc multiples of 16; Nc multiple of 64; K multiple of 32.
// ---------------------------------------------------------------------------
template <bool BIAS, bool RES, bool OUTF_EN, bool OUTB_EN, bool GELU>
__global__ __launch_bounds__(128) void gemm_bf16_kernel(
    const unsigned short* __restrict__ A, const unsigned short* __restrict__ Wb,
    const float* __restrict__ bias, const float* __restrict__ resid,
    float* __restrict__ outF, unsigned short* __restrict__ outB,
    int Mr, int Nc, int K, float scale) {
  int lane = threadIdx.x & 31;
  int wave = threadIdx.x >> 5;
  int ntg = Nc >> 6;                 // groups of 4 n-tiles
  int total = (Mr >> 4) * ntg;
  int tid = blockIdx.x * (blockDim.x >> 5) + wave;
  if (tid >= total) return;
  int tm = tid / ntg, tg = tid % ntg;
  int col16 = lane & 15;
  int half = (lane >> 4) & 1;
  int row = (tm << 4) + col16;
  const unsigned short* Arow = A + (size_t)row * K;
  const unsigned short* Wrow[4];
#pragma unroll
  for (int j = 0; j < 4; ++j)
    Wrow[j] = Wb + (size_t)((tg << 6) + (j << 4) + col16) * K;

  v8f c[4] = {{}, {}, {}, {}};
  for (int k0 = 0; k0 < K; k0 += 32) {
    v16bf a = load_afrag(Arow + k0, half);
#pragma unroll
    for (int j = 0; j < 4; ++j) {
      v16bf b = load_bfrag(Wrow[j] + k0, half);
      c[j] = __builtin_amdgcn_wmma_f32_16x16x32_bf16(false, a, false, b,
                                                     (short)0, c[j], false,
                                                     false);
    }
  }
  int rowbase = (tm << 4) + half * 8;
#pragma unroll
  for (int j = 0; j < 4; ++j) {
    int coln = (tg << 6) + (j << 4) + col16;
    float bv = BIAS ? bias[coln] : 0.f;
#pragma unroll
    for (int r = 0; r < 8; ++r) {
      float v = (c[j][r] + bv) * scale;
      if (GELU) v = 0.5f * v * (1.0f + erff(v * 0.70710678118654752f));
      size_t oidx = (size_t)(rowbase + r) * Nc + coln;
      if (RES) v += resid[oidx];
      if (OUTF_EN) outF[oidx] = v;
      if (OUTB_EN) outB[oidx] = f2bf(v);
    }
  }
}

// ---------------------------------------------------------------------------
// KV projection GEMM: kv = ctx @ mkv_w^T + mkv_b, Nc=512, K=256.
// cols [0,256)  -> kb  bf16 [BM, 256]          (row-major K)
// cols [256,512)-> vT  bf16 [B, HEADS, HD, M]  (transposed V, packed stores)
// ---------------------------------------------------------------------------
__global__ __launch_bounds__(128) void gemm_kv_kernel(
    const unsigned short* __restrict__ A, const unsigned short* __restrict__ Wb,
    const float* __restrict__ bias, unsigned short* __restrict__ kb,
    unsigned short* __restrict__ vT) {
  const int K = kDIM, Nc = 2 * kDIM;
  int lane = threadIdx.x & 31;
  int wave = threadIdx.x >> 5;
  int ntg = Nc >> 6;                  // 8
  int tid = blockIdx.x * (blockDim.x >> 5) + wave;
  int tm = tid / ntg, tg = tid % ntg;
  int col16 = lane & 15;
  int half = (lane >> 4) & 1;
  int row = (tm << 4) + col16;
  const unsigned short* Arow = A + (size_t)row * K;
  const unsigned short* Wrow[4];
#pragma unroll
  for (int j = 0; j < 4; ++j)
    Wrow[j] = Wb + (size_t)((tg << 6) + (j << 4) + col16) * K;

  v8f c[4] = {{}, {}, {}, {}};
  for (int k0 = 0; k0 < K; k0 += 32) {
    v16bf a = load_afrag(Arow + k0, half);
#pragma unroll
    for (int j = 0; j < 4; ++j) {
      v16bf b = load_bfrag(Wrow[j] + k0, half);
      c[j] = __builtin_amdgcn_wmma_f32_16x16x32_bf16(false, a, false, b,
                                                     (short)0, c[j], false,
                                                     false);
    }
  }
  int rowbase = (tm << 4) + half * 8;
  int bidx = rowbase >> 14;           // batch (kM = 16384 rows per batch)
  int mloc = rowbase & (kM - 1);
  if (tg < 4) {                       // K half: row-major [BM, 256]
#pragma unroll
    for (int j = 0; j < 4; ++j) {
      int coln = (tg << 6) + (j << 4) + col16;
      float bv = bias[coln];
#pragma unroll
      for (int r = 0; r < 8; ++r)
        kb[(size_t)(rowbase + r) * kDIM + coln] = f2bf(c[j][r] + bv);
    }
  } else {                            // V half: transposed [B,H,HD,M]
#pragma unroll
    for (int j = 0; j < 4; ++j) {
      int dcol = ((tg - 4) << 6) + (j << 4) + col16;   // 0..255
      int h = dcol >> 5, dd = dcol & 31;
      float bv = bias[256 + dcol];
      unsigned p[4];
#pragma unroll
      for (int r = 0; r < 4; ++r) {
        unsigned lo = f2bf(c[j][2 * r] + bv);
        unsigned hi = f2bf(c[j][2 * r + 1] + bv);
        p[r] = lo | (hi << 16);
      }
      uint4 st = {p[0], p[1], p[2], p[3]};
      size_t vbase = (((size_t)bidx * kHEADS + h) * kHD + dd) * kM + mloc;
      *reinterpret_cast<uint4*>(vT + vbase) = st;
    }
  }
}

// ---------------------------------------------------------------------------
// Masked cross-attention, flash style.
// Grid: B*HEADS*7; 4 waves/block, each owns a 4096-wide m-stripe with online
// softmax. P is re-laid-out through wave-private LDS (no block barriers in
// the loop: same-wave LDS ops are in-order; s_wait_dscnt + wave_barrier
// provide the needed ordering). Partials merged in LDS at the end.
// ---------------------------------------------------------------------------
__global__ __launch_bounds__(128) void cross_attn_kernel(
    const unsigned short* __restrict__ qs,   // [BN, DIM] bf16, pre-scaled
    const unsigned short* __restrict__ kb,   // [BM, 256] bf16
    const unsigned short* __restrict__ vT,   // [B,H,HD,M] bf16
    const unsigned* __restrict__ mwords,     // [B,N,M/32] packed mask bits
    unsigned short* __restrict__ oB) {       // [BN, DIM] bf16
  __shared__ unsigned short lds_p[4][16 * 32];
  __shared__ float lds_acc[4][16 * 32];
  __shared__ float lds_m[4][16];
  __shared__ float lds_l[4][16];

  int lane = threadIdx.x & 31;
  int wave = threadIdx.x >> 5;
  int blk = blockIdx.x;               // b*56 + h*7 + nt
  int nt = blk % 7;
  int h = (blk / 7) % kHEADS;
  int b = blk / (kHEADS * 7);
  int col16 = lane & 15;
  int half = (lane >> 4) & 1;
  int rowbase = nt * 16 + half * 8;
  const float NEGINF = -__builtin_inff();

  // Q fragment (A layout); rows >= N zero
  v16bf qa = {};
  {
    int n = nt * 16 + col16;
    if (n < kN)
      qa = load_afrag(qs + (size_t)(b * kN + n) * kDIM + h * kHD, half);
  }

  float m_r[8], l_r[8];
  v8f acc0 = {}, acc1 = {};
#pragma unroll
  for (int r = 0; r < 8; ++r) { m_r[r] = NEGINF; l_r[r] = 0.f; }

  size_t kbase = (size_t)b * kM;
  size_t vbase0 = ((size_t)b * kHEADS + h) * kHD * (size_t)kM;
  int m_start = wave * (kM / 4);
  for (int m0 = m_start; m0 < m_start + kM / 4; m0 += 32) {
    // ---- scores S = Q @ K^T ----
    v8f s01[2];
#pragma unroll
    for (int mt = 0; mt < 2; ++mt) {
      int mcol = m0 + mt * 16 + col16;
      v16bf bk = load_bfrag(kb + (kbase + mcol) * kDIM + h * kHD, half);
      v8f z = {};
      s01[mt] = __builtin_amdgcn_wmma_f32_16x16x32_bf16(false, qa, false, bk,
                                                        (short)0, z, false,
                                                        false);
    }
    // ---- packed mask word per row (covers all 32 m of this chunk) ----
    unsigned mword[8];
#pragma unroll
    for (int r = 0; r < 8; ++r) {
      int n = rowbase + r;
      mword[r] = (n < kN) ? mwords[(size_t)(b * kN + n) * kMW + (m0 >> 5)] : 0u;
    }
    // ---- mask + per-row max ----
    float s[2][8], mx[8];
#pragma unroll
    for (int r = 0; r < 8; ++r) {
      float best = NEGINF;
#pragma unroll
      for (int mt = 0; mt < 2; ++mt) {
        bool ok = (mword[r] >> (mt * 16 + col16)) & 1u;
        float sv = ok ? s01[mt][r] : NEGINF;
        s[mt][r] = sv;
        best = fmaxf(best, sv);
      }
      mx[r] = best;
    }
#pragma unroll
    for (int r = 0; r < 8; ++r) {
      float v = mx[r];
#pragma unroll
      for (int msk = 8; msk; msk >>= 1) v = fmaxf(v, __shfl_xor(v, msk, 32));
      mx[r] = v;
    }
    float alpha[8];
#pragma unroll
    for (int r = 0; r < 8; ++r) {
      float mnew = fmaxf(m_r[r], mx[r]);
      alpha[r] = (m_r[r] == NEGINF) ? 0.f : __expf(m_r[r] - mnew);
      m_r[r] = mnew;
    }
    // ---- P = exp(S - m) -> wave-private LDS (bf16), row sums ----
    float rs[8];
#pragma unroll
    for (int r = 0; r < 8; ++r) {
      float mnew = m_r[r];
      float sum = 0.f;
#pragma unroll
      for (int mt = 0; mt < 2; ++mt) {
        float p = (s[mt][r] == NEGINF) ? 0.f : __expf(s[mt][r] - mnew);
        sum += p;
        lds_p[wave][(half * 8 + r) * 32 + mt * 16 + col16] = f2bf(p);
      }
      rs[r] = sum;
    }
#pragma unroll
    for (int r = 0; r < 8; ++r) {
      float v = rs[r];
#pragma unroll
      for (int msk = 8; msk; msk >>= 1) v += __shfl_xor(v, msk, 32);
      l_r[r] = l_r[r] * alpha[r] + v;
    }
#pragma unroll
    for (int r = 0; r < 8; ++r) { acc0[r] *= alpha[r]; acc1[r] *= alpha[r]; }
    // same-wave LDS RAW: DS ops are in-order per wave; drain DScnt and stop
    // the compiler from reordering the loads above the stores.
    asm volatile("s_wait_dscnt 0x0" ::: "memory");
    // ---- P as A fragment (b128 from LDS) ----
    const uint4* lp = reinterpret_cast<const uint4*>(&lds_p[wave][col16 * 32]);
    v16bf pa = make_v16bf(lp[half], lp[2 + half]);
    __builtin_amdgcn_wave_barrier();  // keep next-iter stores below these loads
    // ---- O += P @ V (V^T rows are contiguous in m) ----
#pragma unroll
    for (int dt = 0; dt < 2; ++dt) {
      int dd = dt * 16 + col16;
      v16bf bv = load_bfrag(vT + vbase0 + (size_t)dd * kM + m0, half);
      if (dt == 0)
        acc0 = __builtin_amdgcn_wmma_f32_16x16x32_bf16(false, pa, false, bv,
                                                       (short)0, acc0, false,
                                                       false);
      else
        acc1 = __builtin_amdgcn_wmma_f32_16x16x32_bf16(false, pa, false, bv,
                                                       (short)0, acc1, false,
                                                       false);
    }
  }

  // ---- merge wave partials ----
#pragma unroll
  for (int r = 0; r < 8; ++r) {
    lds_acc[wave][(half * 8 + r) * 32 + col16] = acc0[r];
    lds_acc[wave][(half * 8 + r) * 32 + 16 + col16] = acc1[r];
  }
  if (col16 == 0) {
#pragma unroll
    for (int r = 0; r < 8; ++r) {
      lds_m[wave][half * 8 + r] = m_r[r];
      lds_l[wave][half * 8 + r] = l_r[r];
    }
  }
  __syncthreads();
  for (int idx = threadIdx.x; idx < 16 * 32; idx += 128) {
    int n_local = idx >> 5;
    int d = idx & 31;
    int n = nt * 16 + n_local;
    if (n >= kN) continue;
    float mm = NEGINF;
#pragma unroll
    for (int w = 0; w < 4; ++w) mm = fmaxf(mm, lds_m[w][n_local]);
    float L = 0.f, O = 0.f;
#pragma unroll
    for (int w = 0; w < 4; ++w) {
      float sc = (lds_m[w][n_local] == NEGINF) ? 0.f
                                               : __expf(lds_m[w][n_local] - mm);
      L += sc * lds_l[w][n_local];
      O += sc * lds_acc[w][n_local * 32 + d];
    }
    float o = (L > 0.f) ? O / L : 0.f;
    oB[(size_t)(b * kN + n) * kDIM + h * kHD + d] = f2bf(o);
  }
}

// ---------------------------------------------------------------------------
// Self-attention: seq = b (8), batch = n (100). One wave per (n,h), lane = d.
// ---------------------------------------------------------------------------
__global__ __launch_bounds__(128) void self_attn_kernel(
    const float* __restrict__ qkv, unsigned short* __restrict__ soB) {
  int lane = threadIdx.x & 31;
  int wave = threadIdx.x >> 5;
  int id = blockIdx.x * (blockDim.x >> 5) + wave;  // n*8 + h
  if (id >= kN * kHEADS) return;
  int h = id & 7;
  int n = id >> 3;
  float kk[8], vv[8];
#pragma unroll
  for (int j = 0; j < kB; ++j) {
    size_t rowj = (size_t)(j * kN + n) * (3 * kDIM);
    kk[j] = qkv[rowj + kDIM + h * kHD + lane];
    vv[j] = qkv[rowj + 2 * kDIM + h * kHD + lane];
  }
#pragma unroll
  for (int i = 0; i < kB; ++i) {
    size_t rowi = (size_t)(i * kN + n) * (3 * kDIM);
    float q = qkv[rowi + h * kHD + lane] * kSCALE;
    float sc[8];
    float mx = -__builtin_inff();
#pragma unroll
    for (int j = 0; j < kB; ++j) {
      float p = q * kk[j];
#pragma unroll
      for (int msk = 16; msk; msk >>= 1) p += __shfl_xor(p, msk, 32);
      sc[j] = p;
      mx = fmaxf(mx, p);
    }
    float sum = 0.f;
#pragma unroll
    for (int j = 0; j < kB; ++j) { sc[j] = __expf(sc[j] - mx); sum += sc[j]; }
    float inv = 1.f / sum;
    float o = 0.f;
#pragma unroll
    for (int j = 0; j < kB; ++j) o += sc[j] * vv[j];
    o *= inv;
    soB[(size_t)(i * kN + n) * kDIM + h * kHD + lane] = f2bf(o);
  }
}

// ---------------------------------------------------------------------------
// Host side
// ---------------------------------------------------------------------------
extern "C" void kernel_launch(void* const* d_in, const int* in_sizes, int n_in,
                              void* d_out, int out_size, void* d_ws,
                              size_t ws_size, hipStream_t stream) {
  (void)in_sizes; (void)n_in; (void)out_size; (void)ws_size;
  const float* x       = (const float*)d_in[0];
  const float* ctx     = (const float*)d_in[1];
  const int*   mask    = (const int*)d_in[2];
  const float* mq_w    = (const float*)d_in[3];
  const float* mq_b    = (const float*)d_in[4];
  const float* mkv_w   = (const float*)d_in[5];
  const float* mkv_b   = (const float*)d_in[6];
  const float* mproj_w = (const float*)d_in[7];
  const float* mproj_b = (const float*)d_in[8];
  const float* in_w    = (const float*)d_in[9];
  const float* in_b    = (const float*)d_in[10];
  const float* out_w   = (const float*)d_in[11];
  const float* out_b   = (const float*)d_in[12];
  const float* w1      = (const float*)d_in[13];
  const float* b1      = (const float*)d_in[14];
  const float* w2      = (const float*)d_in[15];
  const float* b2      = (const float*)d_in[16];
  const float* n1_g    = (const float*)d_in[17];
  const float* n1_b    = (const float*)d_in[18];
  const float* n2_g    = (const float*)d_in[19];
  const float* n2_b    = (const float*)d_in[20];
  const float* n3_g    = (const float*)d_in[21];
  const float* n3_b    = (const float*)d_in[22];
  const float* n4_g    = (const float*)d_in[23];
  const float* n4_b    = (const float*)d_in[24];
  float* out = (float*)d_out;

  char* ws = (char*)d_ws;
  size_t off = 0;
  auto alloc = [&](size_t bytes) -> void* {
    off = (off + 255) & ~(size_t)255;
    void* p = ws + off;
    off += bytes;
    return p;
  };

  unsigned short* ctxb  = (unsigned short*)alloc((size_t)kBM * kDIM * 2);
  unsigned short* kbuf  = (unsigned short*)alloc((size_t)kBM * kDIM * 2);
  unsigned short* vT    = (unsigned short*)alloc((size_t)kBM * kDIM * 2);
  unsigned*       mw    = (unsigned*)alloc((size_t)kBN * kMW * 4);
  unsigned short* y1b   = (unsigned short*)alloc((size_t)kBN * kDIM * 2);
  unsigned short* q_bf  = (unsigned short*)alloc((size_t)kBN * kDIM * 2);
  unsigned short* o_bf  = (unsigned short*)alloc((size_t)kBN * kDIM * 2);
  float*          x1    = (float*)alloc((size_t)kBN * kDIM * 4);
  unsigned short* y2b   = (unsigned short*)alloc((size_t)kBN * kDIM * 2);
  float*          qkvb  = (float*)alloc((size_t)kBN * 3 * kDIM * 4);
  unsigned short* so_bf = (unsigned short*)alloc((size_t)kBN * kDIM * 2);
  float*          x2    = (float*)alloc((size_t)kBN * kDIM * 4);
  unsigned short* y3b   = (unsigned short*)alloc((size_t)kBN * kDIM * 2);
  unsigned short* f_bf  = (unsigned short*)alloc((size_t)kBN * kFFN * 2);
  float*          x3    = (float*)alloc((size_t)kBN * kDIM * 4);
  // bf16 weights
  unsigned short* mq_wb    = (unsigned short*)alloc((size_t)kDIM * kDIM * 2);
  unsigned short* mkv_wb   = (unsigned short*)alloc((size_t)2 * kDIM * kDIM * 2);
  unsigned short* mproj_wb = (unsigned short*)alloc((size_t)kDIM * kDIM * 2);
  unsigned short* in_wb    = (unsigned short*)alloc((size_t)3 * kDIM * kDIM * 2);
  unsigned short* out_wb   = (unsigned short*)alloc((size_t)kDIM * kDIM * 2);
  unsigned short* w1b      = (unsigned short*)alloc((size_t)kFFN * kDIM * 2);
  unsigned short* w2b      = (unsigned short*)alloc((size_t)kDIM * kFFN * 2);

  auto cvt = [&](const float* src, unsigned short* dst, size_t n) {
    int blocks = (int)((n + 255 * 8) / (256 * 8));
    if (blocks > 8192) blocks = 8192;
    if (blocks < 1) blocks = 1;
    cvt_bf16_kernel<<<blocks, 256, 0, stream>>>(src, dst, n);
  };

  // conversions + mask packing
  cvt(ctx, ctxb, (size_t)kBM * kDIM);
  cvt(mq_w, mq_wb, (size_t)kDIM * kDIM);
  cvt(mkv_w, mkv_wb, (size_t)2 * kDIM * kDIM);
  cvt(mproj_w, mproj_wb, (size_t)kDIM * kDIM);
  cvt(in_w, in_wb, (size_t)3 * kDIM * kDIM);
  cvt(out_w, out_wb, (size_t)kDIM * kDIM);
  cvt(w1, w1b, (size_t)kFFN * kDIM);
  cvt(w2, w2b, (size_t)kDIM * kFFN);
  pack_mask_kernel<<<800, 256, 0, stream>>>(mask, mw, kBN * kMW);

  auto blocks_for = [](int Mr, int Nc) { return ((Mr >> 4) * (Nc >> 6) + 3) / 4; };

  // LN1 -> y1b
  ln_kernel<<<(kBN + 3) / 4, 128, 0, stream>>>(x, n1_g, n1_b, nullptr, y1b, kBN);
  // Q = (y1 @ mq_w^T + mq_b) * scale  (bf16)
  gemm_bf16_kernel<true, false, false, true, false>
      <<<blocks_for(kBN, kDIM), 128, 0, stream>>>(
          y1b, mq_wb, mq_b, nullptr, nullptr, q_bf, kBN, kDIM, kDIM, kSCALE);
  // KV projection -> kbuf + vT
  gemm_kv_kernel<<<blocks_for(kBM, 2 * kDIM), 128, 0, stream>>>(
      ctxb, mkv_wb, mkv_b, kbuf, vT);
  // masked cross-attention
  cross_attn_kernel<<<kB * kHEADS * 7, 128, 0, stream>>>(q_bf, kbuf, vT, mw,
                                                         o_bf);
  // x1 = x + o @ mproj_w^T + mproj_b
  gemm_bf16_kernel<true, true, true, false, false>
      <<<blocks_for(kBN, kDIM), 128, 0, stream>>>(
          o_bf, mproj_wb, mproj_b, x, x1, nullptr, kBN, kDIM, kDIM, 1.f);
  // LN2 -> y2b
  ln_kernel<<<(kBN + 3) / 4, 128, 0, stream>>>(x1, n2_g, n2_b, nullptr, y2b, kBN);
  // qkv = y2 @ in_w^T + in_b (f32)
  gemm_bf16_kernel<true, false, true, false, false>
      <<<blocks_for(kBN, 3 * kDIM), 128, 0, stream>>>(
          y2b, in_wb, in_b, nullptr, qkvb, nullptr, kBN, 3 * kDIM, kDIM, 1.f);
  // self-attention
  self_attn_kernel<<<(kN * kHEADS + 3) / 4, 128, 0, stream>>>(qkvb, so_bf);
  // x2 = x1 + so @ out_w^T + out_b
  gemm_bf16_kernel<true, true, true, false, false>
      <<<blocks_for(kBN, kDIM), 128, 0, stream>>>(
          so_bf, out_wb, out_b, x1, x2, nullptr, kBN, kDIM, kDIM, 1.f);
  // LN3 -> y3b
  ln_kernel<<<(kBN + 3) / 4, 128, 0, stream>>>(x2, n3_g, n3_b, nullptr, y3b, kBN);
  // f = gelu(y3 @ w1^T + b1) (bf16)
  gemm_bf16_kernel<true, false, false, true, true>
      <<<blocks_for(kBN, kFFN), 128, 0, stream>>>(
          y3b, w1b, b1, nullptr, nullptr, f_bf, kBN, kFFN, kDIM, 1.f);
  // x3 = x2 + f @ w2^T + b2
  gemm_bf16_kernel<true, true, true, false, false>
      <<<blocks_for(kBN, kDIM), 128, 0, stream>>>(
          f_bf, w2b, b2, x2, x3, nullptr, kBN, kDIM, kFFN, 1.f);
  // out = LN4(x3)
  ln_kernel<<<(kBN + 3) / 4, 128, 0, stream>>>(x3, n4_g, n4_b, out, nullptr, kBN);
}
